// EncoderBlock_33088428048644
// MI455X (gfx1250) — matrix operand order repaired and checked by reading, not compiled, not run on hardware
//
#include <hip/hip_runtime.h>
#include <cstdint>

typedef __attribute__((ext_vector_type(16))) _Float16 v16h;
typedef __attribute__((ext_vector_type(4)))  _Float16 v4h;
typedef __attribute__((ext_vector_type(8)))  float    v8f;
typedef __attribute__((ext_vector_type(4)))  float    v4f;

#define WMMA16(a, b, c) \
  __builtin_amdgcn_wmma_f32_16x16x32_f16(false, (a), false, (b), (short)0, (c), false, false)

// ---------------------------------------------------------------------------
// CDNA5 async copy: global -> LDS, 16B per lane, tracked by ASYNCcnt.
__device__ __forceinline__ void async_b128(void* lds, const void* gmem) {
  asm volatile("global_load_async_to_lds_b128 %0, %1, off"
               :: "v"((uint32_t)(uintptr_t)lds),
                  "v"((uint64_t)(uintptr_t)gmem)
               : "memory");
}
__device__ __forceinline__ void wait_async0() {
  asm volatile("s_wait_asynccnt 0" ::: "memory");
}

// ---------------------------------------------------------------------------
// Fragment loaders (wave32, v_wmma_f32_16x16x32_f16).
// A 16x32 (ISA 7.12.2): lane m/group g; two contiguous 8-half runs at 8g, 16+8g.
__device__ __forceinline__ v16h lds_frag_a(const _Float16* p, int ld) {
  int lane = threadIdx.x & 31;
  int m = lane & 15, g = lane >> 4;
  const _Float16* row = p + m * ld + 8 * g;
  v16h f;
#pragma unroll
  for (int i = 0; i < 8; ++i) f[i] = row[i];
#pragma unroll
  for (int i = 0; i < 8; ++i) f[8 + i] = row[16 + i];
  return f;
}
// B 32x16: lane group g holds K=8g..8g+7 and 16+8g..16+8g+7 of column n; tiles
// stored transposed in LDS ([n][k]) -> two contiguous 8-half runs.
__device__ __forceinline__ v16h lds_frag_b(const _Float16* p, int ld) {
  int lane = threadIdx.x & 31;
  int n = lane & 15, g = lane >> 4;
  const _Float16* row = p + n * ld + 8 * g;
  v16h f;
#pragma unroll
  for (int i = 0; i < 8; ++i) f[i] = row[i];
#pragma unroll
  for (int i = 0; i < 8; ++i) f[8 + i] = row[16 + i];
  return f;
}

// ---------------------------------------------------------------------------
// f32 -> f16 conversion, 4 elements/thread
__global__ __launch_bounds__(256) void cvt_h(const float* __restrict__ in,
                                             _Float16* __restrict__ out, int n) {
  int i = (blockIdx.x * 256 + threadIdx.x) * 4;
  if (i < n) {
    v4f d = *(const v4f*)(in + i);
    v4h o;
#pragma unroll
    for (int j = 0; j < 4; ++j) o[j] = (_Float16)d[j];
    *(v4h*)(out + i) = o;
  }
}

// f32 [K,N] -> f16 [N,K] transpose-convert (32x32 LDS tiles, grid.z = head)
__global__ __launch_bounds__(256)
void cvt_h_t(const float* __restrict__ in, _Float16* __restrict__ out,
             int K, int N) {
  __shared__ float tile[32][33];
  size_t hoff = (size_t)blockIdx.z * K * N;
  int k0 = blockIdx.y * 32, n0 = blockIdx.x * 32;
  int tx = threadIdx.x & 31, ty = threadIdx.x >> 5;
#pragma unroll
  for (int i = 0; i < 32; i += 8)
    tile[ty + i][tx] = in[hoff + (size_t)(k0 + ty + i) * N + n0 + tx];
  __syncthreads();
#pragma unroll
  for (int i = 0; i < 32; i += 8)
    out[hoff + (size_t)(n0 + ty + i) * K + k0 + tx] = (_Float16)tile[tx][ty + i];
}

// ---------------------------------------------------------------------------
// Tiled WMMA GEMM: C[M,N] = A[M,K]*BT[N,K]^T + bias; optional ReLU, output
// scale, and transposed f16 output. 256 threads = 8 waves (4x2 of 32x32);
// BM=128 BN=64 BK=32; double-buffered LDS; ALL staging is async-to-LDS b128.
#define BM 128
#define BN 64
#define BK 32
#define LDSA 40
#define LDSB 40

__global__ __launch_bounds__(256)
void gemm_wmma(const _Float16* __restrict__ A, int lda,
               const _Float16* __restrict__ BT, int ldb, long sB,
               const float* __restrict__ bias, int sBias,
               float* __restrict__ Cf, int ldc, long sCf,
               _Float16* __restrict__ Ch, int ldh, long sCh, int tposeCh,
               int M, int N, int K, int relu, float oscale) {
  __shared__ _Float16 As[2][BM * LDSA];
  __shared__ _Float16 Bt[2][BN * LDSB];

  int head = blockIdx.y;
  int mtiles = M / BM;
  int bm = blockIdx.x % mtiles;
  int bn = blockIdx.x / mtiles;
  int tid = threadIdx.x;
  int wid = tid >> 5, lane = tid & 31;
  int wm = wid >> 1, wn = wid & 1;
  int n = lane & 15, g = lane >> 4;

  // staging geometry: A = 512 16B chunks (2/thread), B = 256 16B chunks (1/thread)
  const int arow = tid >> 2, akc = (tid & 3) * 8;
  const int bnn = tid >> 2, bkc = (tid & 3) * 8;
  const _Float16* Ag = A + (size_t)bm * BM * lda;
  const _Float16* Bg = BT + (size_t)head * sB + (size_t)bn * BN * ldb;

  auto stage = [&](int bb, int kt) {
    async_b128(&As[bb][arow * LDSA + akc], Ag + (size_t)arow * lda + kt + akc);
    async_b128(&As[bb][(arow + 64) * LDSA + akc],
               Ag + (size_t)(arow + 64) * lda + kt + akc);
    async_b128(&Bt[bb][bnn * LDSB + bkc], Bg + (size_t)bnn * ldb + kt + bkc);
  };

  const v8f vzero = {0.f, 0.f, 0.f, 0.f, 0.f, 0.f, 0.f, 0.f};
  v8f acc[2][2];
#pragma unroll
  for (int i = 0; i < 2; ++i)
#pragma unroll
    for (int j = 0; j < 2; ++j) acc[i][j] = vzero;

  stage(0, 0);
  wait_async0();
  __syncthreads();

  int bb = 0;
  for (int kt = 0; kt < K; kt += BK) {
    const _Float16* Ab = &As[bb][(wm * 32) * LDSA];
    const _Float16* Bb = &Bt[bb][(wn * 32) * LDSB];
    v16h a0 = lds_frag_a(Ab, LDSA);
    v16h a1 = lds_frag_a(Ab + 16 * LDSA, LDSA);
    v16h b0 = lds_frag_b(Bb, LDSB);
    v16h b1 = lds_frag_b(Bb + 16 * LDSB, LDSB);

    bool more = (kt + BK) < K;
    if (more) stage(bb ^ 1, kt + BK);  // overlap next tile with WMMAs

    acc[0][0] = WMMA16(a0, b0, acc[0][0]);
    acc[0][1] = WMMA16(a0, b1, acc[0][1]);
    acc[1][0] = WMMA16(a1, b0, acc[1][0]);
    acc[1][1] = WMMA16(a1, b1, acc[1][1]);

    if (more) {
      wait_async0();
      __syncthreads();
      bb ^= 1;
    }
  }

  const float* biash = bias + (size_t)head * sBias;
#pragma unroll
  for (int i = 0; i < 2; ++i)
#pragma unroll
    for (int j = 0; j < 2; ++j) {
      int row0 = bm * BM + wm * 32 + i * 16 + 8 * g;
      int col = bn * BN + wn * 32 + j * 16 + n;
      float bv = biash[col];
#pragma unroll
      for (int r = 0; r < 8; ++r) {
        float v = (acc[i][j][r] + bv) * oscale;
        if (relu) v = fmaxf(v, 0.f);
        if (Cf) Cf[(size_t)head * sCf + (size_t)(row0 + r) * ldc + col] = v;
        if (Ch) {
          size_t idx = tposeCh ? (size_t)head * sCh + (size_t)col * ldh + (row0 + r)
                               : (size_t)head * sCh + (size_t)(row0 + r) * ldh + col;
          Ch[idx] = (_Float16)v;
        }
      }
    }
}

// ---------------------------------------------------------------------------
// Flash attention. q/k: [H, B*S, 64] f16 (q pre-scaled by Dh^-0.5 in its GEMM);
// v: TRANSPOSED [H, 64, B*S] f16. All tiles staged via async-to-LDS b128,
// K/V double-buffered. Writes attn [B, S, D] f32 (head transpose fused).
#define AS 2048
#define AMR 4096
#define AD 1024
#define ADh 64
#define ALD 72

__global__ __launch_bounds__(128)
void flash_attn(const _Float16* __restrict__ q, const _Float16* __restrict__ k,
                const _Float16* __restrict__ vT, const int* __restrict__ mask,
                float* __restrict__ attn) {
  __shared__ _Float16 Qs[64 * ALD];
  __shared__ _Float16 Ks[2][64 * ALD];   // [key][dim]
  __shared__ _Float16 Vt[2][64 * ALD];   // [dim][key]
  __shared__ _Float16 Ps[4 * 16 * ALD];
  __shared__ int Ms[2][64];

  int q0 = blockIdx.x * 64;
  int by = blockIdx.y;  // h*B + b
  int h = by >> 1, b = by & 1;
  size_t rowbase = (size_t)by * AS;
  const _Float16* Qg = q + (rowbase + q0) * ADh;
  const _Float16* Kg = k + rowbase * ADh;
  const _Float16* Vg = vT + (size_t)h * ADh * AMR + (size_t)b * AS;
  const int* Mg = mask + (size_t)b * AS;

  int tid = threadIdx.x, wid = tid >> 5, lane = tid & 31;
  int n = lane & 15, g = lane >> 4;

  auto stageKV = [&](int bb, int t0) {
#pragma unroll
    for (int it = 0; it < 4; ++it) {
      int c = tid + it * 128;
      int row = c >> 3, nc = (c & 7) * 8;
      async_b128(&Ks[bb][row * ALD + nc], Kg + (size_t)(t0 + row) * ADh + nc);
      async_b128(&Vt[bb][row * ALD + nc], Vg + (size_t)row * AMR + t0 + nc);
    }
    Ms[bb][tid & 63] = Mg[t0 + (tid & 63)];
  };

  // Q tile (already scaled): straight async copy
#pragma unroll
  for (int it = 0; it < 4; ++it) {
    int c = tid + it * 128;
    int row = c >> 3, nc = (c & 7) * 8;
    async_b128(&Qs[row * ALD + nc], Qg + (size_t)row * ADh + nc);
  }
  stageKV(0, 0);
  wait_async0();
  __syncthreads();

  v16h aq0 = lds_frag_a(&Qs[(wid * 16) * ALD], ALD);
  v16h aq1 = lds_frag_a(&Qs[(wid * 16) * ALD + 32], ALD);

  const v8f vzero = {0.f, 0.f, 0.f, 0.f, 0.f, 0.f, 0.f, 0.f};
  float mr[8], lr[8];
  v8f o[4];
#pragma unroll
  for (int r = 0; r < 8; ++r) { mr[r] = -1e30f; lr[r] = 0.f; }
#pragma unroll
  for (int j = 0; j < 4; ++j) o[j] = vzero;

  int bb = 0;
  for (int t0 = 0; t0 < AS; t0 += 64) {
    // S = Q K^T (16 rows x 64 keys per wave)
    v8f s[4];
#pragma unroll
    for (int j = 0; j < 4; ++j) {
      v16h b0 = lds_frag_b(&Ks[bb][(j * 16) * ALD], ALD);
      v16h b1 = lds_frag_b(&Ks[bb][(j * 16) * ALD + 32], ALD);
      v8f t = WMMA16(aq0, b0, vzero);
      s[j] = WMMA16(aq1, b1, t);
      if (Ms[bb][j * 16 + n] == 0) {
#pragma unroll
        for (int r = 0; r < 8; ++r) s[j][r] = -1e30f;
      }
    }

    bool more = (t0 + 64) < AS;
    if (more) stageKV(bb ^ 1, t0 + 64);  // overlap next K/V with softmax+PV

    // online softmax; rows live in-lane, reduce across the 16-lane group
    float mnew[8], al[8], rs[8];
#pragma unroll
    for (int r = 0; r < 8; ++r) {
      float t = fmaxf(fmaxf(s[0][r], s[1][r]), fmaxf(s[2][r], s[3][r]));
#pragma unroll
      for (int off = 1; off < 16; off <<= 1) t = fmaxf(t, __shfl_xor(t, off, 32));
      mnew[r] = fmaxf(mr[r], t);
      al[r] = __expf(mr[r] - mnew[r]);
      mr[r] = mnew[r];
      rs[r] = 0.f;
    }
#pragma unroll
    for (int j = 0; j < 4; ++j)
#pragma unroll
      for (int r = 0; r < 8; ++r) {
        float p = __expf(s[j][r] - mnew[r]);
        s[j][r] = p;
        rs[r] += p;
      }
#pragma unroll
    for (int r = 0; r < 8; ++r) {
#pragma unroll
      for (int off = 1; off < 16; off <<= 1) rs[r] += __shfl_xor(rs[r], off, 32);
      lr[r] = lr[r] * al[r] + rs[r];
    }
#pragma unroll
    for (int j = 0; j < 4; ++j)
#pragma unroll
      for (int r = 0; r < 8; ++r) o[j][r] *= al[r];

    // reshape P (C layout) -> A fragment via LDS
    _Float16* Pw = &Ps[(wid * 16) * ALD];
#pragma unroll
    for (int j = 0; j < 4; ++j)
#pragma unroll
      for (int r = 0; r < 8; ++r)
        Pw[(r + 8 * g) * ALD + j * 16 + n] = (_Float16)s[j][r];
    __syncthreads();
    v16h pa0 = lds_frag_a(Pw, ALD);
    v16h pa1 = lds_frag_a(Pw + 32, ALD);

    // O += P V
#pragma unroll
    for (int j = 0; j < 4; ++j) {
      v16h b0 = lds_frag_b(&Vt[bb][(j * 16) * ALD], ALD);
      v16h b1 = lds_frag_b(&Vt[bb][(j * 16) * ALD + 32], ALD);
      o[j] = WMMA16(pa0, b0, o[j]);
      o[j] = WMMA16(pa1, b1, o[j]);
    }

    if (more) {
      wait_async0();
      __syncthreads();
      bb ^= 1;
    }
  }

  // write [B,S,D] with head offset
#pragma unroll
  for (int j = 0; j < 4; ++j)
#pragma unroll
    for (int r = 0; r < 8; ++r) {
      int srow = q0 + wid * 16 + r + 8 * g;
      int col = h * ADh + j * 16 + n;
      attn[((size_t)b * AS + srow) * AD + col] = o[j][r] / lr[r];
    }
}

// ---------------------------------------------------------------------------
// out = res + alpha*(a-mean)/(std_unbiased + eps) + beta, rows of D=1024.
__global__ __launch_bounds__(256)
void ln_residual(const float* __restrict__ a, const float* __restrict__ res,
                 const float* __restrict__ alpha, const float* __restrict__ beta,
                 float* __restrict__ outf, _Float16* __restrict__ outh) {
  const int D = 1024;
  __shared__ float red[16];
  size_t row = blockIdx.x;
  const float* ar = a + row * D;
  const float* rr = res + row * D;
  int tid = threadIdx.x;
  int c4 = tid * 4;

  v4f xv = *(const v4f*)(ar + c4);
  float s = xv[0] + xv[1] + xv[2] + xv[3];
  float ss = xv[0] * xv[0] + xv[1] * xv[1] + xv[2] * xv[2] + xv[3] * xv[3];
#pragma unroll
  for (int off = 16; off; off >>= 1) {
    s += __shfl_xor(s, off, 32);
    ss += __shfl_xor(ss, off, 32);
  }
  int wid = tid >> 5, lane = tid & 31;
  if (lane == 0) { red[wid] = s; red[8 + wid] = ss; }
  __syncthreads();
  if (tid == 0) {
    float ts = 0.f, tss = 0.f;
#pragma unroll
    for (int i = 0; i < 8; ++i) { ts += red[i]; tss += red[8 + i]; }
    red[0] = ts;
    red[1] = tss;
  }
  __syncthreads();
  float mean = red[0] / D;
  float var = (red[1] - (float)D * mean * mean) / (float)(D - 1);
  float rinv = 1.f / (sqrtf(var) + 1e-6f);

  v4f av = *(const v4f*)(alpha + c4);
  v4f bv = *(const v4f*)(beta + c4);
  v4f rv = *(const v4f*)(rr + c4);
  v4f y;
#pragma unroll
  for (int j = 0; j < 4; ++j) y[j] = av[j] * (xv[j] - mean) * rinv + bv[j] + rv[j];
  *(v4f*)(outf + row * D + c4) = y;
  if (outh) {
    v4h yh;
#pragma unroll
    for (int j = 0; j < 4; ++j) yh[j] = (_Float16)y[j];
    *(v4h*)(outh + row * D + c4) = yh;
  }
}

// ---------------------------------------------------------------------------
extern "C" void kernel_launch(void* const* d_in, const int* in_sizes, int n_in,
                              void* d_out, int out_size, void* d_ws, size_t ws_size,
                              hipStream_t stream) {
  (void)in_sizes; (void)n_in; (void)out_size; (void)ws_size;
  const float* x   = (const float*)d_in[0];
  const int*   msk = (const int*)d_in[1];
  const float* Wq  = (const float*)d_in[2];
  const float* bq  = (const float*)d_in[3];
  const float* Wk  = (const float*)d_in[4];
  const float* bk  = (const float*)d_in[5];
  const float* Wv  = (const float*)d_in[6];
  const float* bv  = (const float*)d_in[7];
  const float* W1  = (const float*)d_in[8];
  const float* b1  = (const float*)d_in[9];
  const float* W2  = (const float*)d_in[10];
  const float* b2  = (const float*)d_in[11];
  const float* al1 = (const float*)d_in[12];
  const float* be1 = (const float*)d_in[13];
  const float* al2 = (const float*)d_in[14];
  const float* be2 = (const float*)d_in[15];
  float* out = (float*)d_out;

  const int Bc = 2, Sc = 2048, Dc = 1024, Hc = 16, Dhc = 64;
  const int Mrows = Bc * Sc;  // 4096

  char* w = (char*)d_ws;
  auto take = [&](size_t bytes) {
    char* p = w;
    w += (bytes + 255) & ~(size_t)255;
    return p;
  };
  _Float16* x_h   = (_Float16*)take((size_t)Mrows * Dc * 2);
  _Float16* wq_h  = (_Float16*)take((size_t)Dc * Dc * 2);      // [H, Dh, D]
  _Float16* wk_h  = (_Float16*)take((size_t)Dc * Dc * 2);      // [H, Dh, D]
  _Float16* wv_h  = (_Float16*)take((size_t)Dc * Dc * 2);      // [H, Dh, D]
  _Float16* w1_h  = (_Float16*)take((size_t)Dc * 4 * Dc * 2);  // [4D, D]
  _Float16* w2_h  = (_Float16*)take((size_t)4 * Dc * Dc * 2);  // [D, 4D]
  _Float16* q_h   = (_Float16*)take((size_t)Hc * Mrows * Dhc * 2);  // [H,B*S,Dh]
  _Float16* k_h   = (_Float16*)take((size_t)Hc * Mrows * Dhc * 2);  // [H,B*S,Dh]
  _Float16* vT_h  = (_Float16*)take((size_t)Hc * Mrows * Dhc * 2);  // [H,Dh,B*S]
  float*    att_f = (float*)take((size_t)Mrows * Dc * 4);
  float*    x1_f  = (float*)take((size_t)Mrows * Dc * 4);
  _Float16* x1_h  = (_Float16*)take((size_t)Mrows * Dc * 2);
  _Float16* hh_h  = (_Float16*)take((size_t)Mrows * 4 * Dc * 2);
  float*    ffn_f = (float*)take((size_t)Mrows * Dc * 4);

  // activations: straight f32->f16
  cvt_h<<<(Mrows * Dc / 4 + 255) / 256, 256, 0, stream>>>(x, x_h, Mrows * Dc);
  // weights: transpose-convert once so GEMM B staging is a straight copy
  cvt_h_t<<<dim3(Dhc / 32, Dc / 32, Hc), 256, 0, stream>>>(Wq, wq_h, Dc, Dhc);
  cvt_h_t<<<dim3(Dhc / 32, Dc / 32, Hc), 256, 0, stream>>>(Wk, wk_h, Dc, Dhc);
  cvt_h_t<<<dim3(Dhc / 32, Dc / 32, Hc), 256, 0, stream>>>(Wv, wv_h, Dc, Dhc);
  cvt_h_t<<<dim3(4 * Dc / 32, Dc / 32, 1), 256, 0, stream>>>(W1, w1_h, Dc, 4 * Dc);
  cvt_h_t<<<dim3(Dc / 32, 4 * Dc / 32, 1), 256, 0, stream>>>(W2, w2_h, 4 * Dc, Dc);

  // per-head QKV projections (q pre-scaled by Dh^-0.5; v written transposed)
  dim3 gq(Mrows / BM, Hc);
  gemm_wmma<<<gq, 256, 0, stream>>>(x_h, Dc, wq_h, Dc, (long)Dc * Dhc, bq, Dhc,
                                    (float*)nullptr, 0, 0, q_h, Dhc,
                                    (long)Mrows * Dhc, 0, Mrows, Dhc, Dc, 0,
                                    0.125f);
  gemm_wmma<<<gq, 256, 0, stream>>>(x_h, Dc, wk_h, Dc, (long)Dc * Dhc, bk, Dhc,
                                    (float*)nullptr, 0, 0, k_h, Dhc,
                                    (long)Mrows * Dhc, 0, Mrows, Dhc, Dc, 0,
                                    1.0f);
  gemm_wmma<<<gq, 256, 0, stream>>>(x_h, Dc, wv_h, Dc, (long)Dc * Dhc, bv, Dhc,
                                    (float*)nullptr, 0, 0, vT_h, Mrows,
                                    (long)Dhc * Mrows, 1, Mrows, Dhc, Dc, 0,
                                    1.0f);

  // flash attention
  dim3 ga(Sc / 64, Hc * Bc);
  flash_attn<<<ga, 128, 0, stream>>>(q_h, k_h, vT_h, msk, att_f);

  // x1 = x + LN(attn)
  ln_residual<<<Mrows, 256, 0, stream>>>(att_f, x, al1, be1, x1_f, x1_h);

  // h = relu(x1 @ W1 + b1) -> f16
  dim3 g1((Mrows / BM) * (4 * Dc / BN), 1);
  gemm_wmma<<<g1, 256, 0, stream>>>(x1_h, Dc, w1_h, Dc, 0, b1, 0,
                                    (float*)nullptr, 0, 0, hh_h, 4 * Dc, 0, 0,
                                    Mrows, 4 * Dc, Dc, 1, 1.0f);
  // ffn = h @ W2 + b2 -> f32
  dim3 g2((Mrows / BM) * (Dc / BN), 1);
  gemm_wmma<<<g2, 256, 0, stream>>>(hh_h, 4 * Dc, w2_h, 4 * Dc, 0, b2, 0, ffn_f,
                                    Dc, 0, (_Float16*)nullptr, 0, 0, 0, Mrows,
                                    Dc, 4 * Dc, 0, 1.0f);
  // out = x1 + LN(ffn)
  ln_residual<<<Mrows, 256, 0, stream>>>(ffn_f, x1_f, al2, be2, out,
                                         (_Float16*)nullptr);
}
